// GraphAttentionNetwork_7060926235078
// MI455X (gfx1250) — compile-verified, matrix operand order checked by reading, not run on hardware
//
#include <hip/hip_runtime.h>
#include <hip/hip_bf16.h>
#include <math.h>

// ---------------- types for WMMA ----------------
typedef __bf16 bf16_t;
typedef __attribute__((ext_vector_type(16))) __bf16 v16bf;
typedef __attribute__((ext_vector_type(8)))  float  v8f;

#define NODE_F 256
#define HID    128
#define HEADS  4
#define OUTF   256

// ---------------- small helpers ----------------
static __device__ __forceinline__ unsigned short f2bf_rne(float x) {
    unsigned int u = __float_as_uint(x);
    unsigned int r = (u + 0x7FFFu + ((u >> 16) & 1u)) >> 16;
    return (unsigned short)r;
}

__global__ void fill_f32(float* p, float v, int n) {
    int i = blockIdx.x * blockDim.x + threadIdx.x;
    if (i < n) p[i] = v;
}

__global__ void cvt_f32_bf16(const float* __restrict__ src,
                             unsigned short* __restrict__ dst, int n) {
    int i = blockIdx.x * blockDim.x + threadIdx.x;
    if (i < n) dst[i] = f2bf_rne(src[i]);
}

// W [K x C] fp32 row-major  ->  Wt [C x K] bf16 (transposed)
__global__ void cvt_transpose_bf16(const float* __restrict__ W,
                                   unsigned short* __restrict__ Wt,
                                   int K, int C) {
    int i = blockIdx.x * blockDim.x + threadIdx.x;
    if (i < K * C) {
        int k = i / C, c = i % C;
        Wt[(size_t)c * K + k] = f2bf_rne(W[i]);
    }
}

// ---------------- WMMA GEMM: C[M,Ncols] = A[M,K] * B[K,Ncols] ----------------
// A is bf16 row-major [M x K]; Bt is bf16 [Ncols x K] (i.e. B transposed).
// ISA 16-bit A-frag layout (16x32): lane = 16*half + m;
//   elements 0..7  -> K = k0 + half*8 + i
//   elements 8..15 -> K = k0 + 16 + half*8 + (i-8)
// Both halves are 8 contiguous bf16 => two 16B loads per fragment per lane.
struct alignas(16) bf8x { bf16_t v[8]; };

static __device__ __forceinline__ v16bf load_frag(const bf16_t* __restrict__ rowptr,
                                                  int k0, int half) {
    bf8x lo = *(const bf8x*)(rowptr + k0 + half * 8);
    bf8x hi = *(const bf8x*)(rowptr + k0 + 16 + half * 8);
    v16bf r;
#pragma unroll
    for (int i = 0; i < 8; ++i) { r[i] = lo.v[i]; r[i + 8] = hi.v[i]; }
    return r;
}

// block = 256 threads = 8 waves, arranged 4 (M) x 2 (N); wave tile = 32x32.
// block tile = 128 (M) x 64 (N). Ncols must be a multiple of 64 (512/256 are).
__global__ void gemm_bf16_wmma(const bf16_t* __restrict__ A,
                               const bf16_t* __restrict__ Bt,
                               float* __restrict__ C,
                               int M, int K, int Ncols) {
    const int lane = threadIdx.x & 31;
    const int wave = threadIdx.x >> 5;
    const int half = lane >> 4;
    const int l16  = lane & 15;

    const int rowBase = blockIdx.x * 128 + (wave >> 1) * 32;
    const int colBase = blockIdx.y * 64  + (wave & 1) * 32;

    v8f acc00 = 0.f, acc01 = 0.f, acc10 = 0.f, acc11 = 0.f;

    int rA0 = rowBase + l16;       if (rA0 > M - 1) rA0 = M - 1;
    int rA1 = rowBase + 16 + l16;  if (rA1 > M - 1) rA1 = M - 1;
    const bf16_t* a0p = A + (size_t)rA0 * K;
    const bf16_t* a1p = A + (size_t)rA1 * K;
    const bf16_t* b0p = Bt + (size_t)(colBase + l16) * K;
    const bf16_t* b1p = Bt + (size_t)(colBase + 16 + l16) * K;

    for (int k0 = 0; k0 < K; k0 += 32) {
        v16bf a0 = load_frag(a0p, k0, half);
        v16bf a1 = load_frag(a1p, k0, half);
        v16bf b0 = load_frag(b0p, k0, half);
        v16bf b1 = load_frag(b1p, k0, half);
        acc00 = __builtin_amdgcn_wmma_f32_16x16x32_bf16(false, a0, false, b0, (short)0, acc00, false, false);
        acc01 = __builtin_amdgcn_wmma_f32_16x16x32_bf16(false, a0, false, b1, (short)0, acc01, false, false);
        acc10 = __builtin_amdgcn_wmma_f32_16x16x32_bf16(false, a1, false, b0, (short)0, acc10, false, false);
        acc11 = __builtin_amdgcn_wmma_f32_16x16x32_bf16(false, a1, false, b1, (short)0, acc11, false, false);
    }

    // C/D layout: VGPR r -> M = r + 8*half, N = l16
#pragma unroll
    for (int r = 0; r < 8; ++r) {
        int m0 = rowBase + r + half * 8;
        int m1 = m0 + 16;
        int c0 = colBase + l16;
        int c1 = c0 + 16;
        if (m0 < M) {
            C[(size_t)m0 * Ncols + c0] = acc00[r];
            C[(size_t)m0 * Ncols + c1] = acc01[r];
        }
        if (m1 < M) {
            C[(size_t)m1 * Ncols + c0] = acc10[r];
            C[(size_t)m1 * Ncols + c1] = acc11[r];
        }
    }
}

// ---------------- GAT graph kernels ----------------
// alpha_src[n,h] = dot(Wh[n,h,:], a[h,:F]); alpha_dst[n,h] = dot(Wh[n,h,:], a[h,F:])
__global__ void alpha_kernel(const float* __restrict__ Wh, const float* __restrict__ a,
                             float* __restrict__ as, float* __restrict__ ad,
                             int N, int H, int F) {
    int i = blockIdx.x * blockDim.x + threadIdx.x;
    if (i >= N * H) return;
    int n = i / H, h = i % H;
    const float* w  = Wh + (size_t)n * H * F + (size_t)h * F;
    const float* av = a + (size_t)h * 2 * F;
    float s = 0.f, d = 0.f;
    for (int f = 0; f < F; ++f) { s += w[f] * av[f]; d += w[f] * av[F + f]; }
    as[i] = s; ad[i] = d;
}

static __device__ __forceinline__ void atomicMaxF(float* addr, float val) {
    unsigned int* ua = (unsigned int*)addr;
    unsigned int old = *ua;
    while (__uint_as_float(old) < val) {
        unsigned int assumed = old;
        old = atomicCAS(ua, assumed, __float_as_uint(val));
        if (old == assumed) break;
    }
}

// per-edge leaky-relu score, scatter-max into padded[dst,h]
__global__ void score_kernel(const int* __restrict__ src, const int* __restrict__ dst,
                             const float* __restrict__ as, const float* __restrict__ ad,
                             float* __restrict__ padded, int E, int H) {
    int i = blockIdx.x * blockDim.x + threadIdx.x;
    if (i >= E * H) return;
    int e = i / H, h = i % H;
    int s = src[e], d = dst[e];
    float v = as[s * H + h] + ad[d * H + h];
    v = v > 0.f ? v : 0.2f * v;
    atomicMaxF(&padded[d * H + h], v);
}

__global__ void head_max(const float* __restrict__ padded, float* __restrict__ hmax,
                         int N, int H) {
    __shared__ float sd[256];
    int h = blockIdx.x;
    float m = -INFINITY;
    for (int n = threadIdx.x; n < N; n += blockDim.x) m = fmaxf(m, padded[n * H + h]);
    sd[threadIdx.x] = m; __syncthreads();
    for (int s = 128; s > 0; s >>= 1) {
        if (threadIdx.x < s) sd[threadIdx.x] = fmaxf(sd[threadIdx.x], sd[threadIdx.x + s]);
        __syncthreads();
    }
    if (threadIdx.x == 0) hmax[h] = sd[0];
}

__global__ void head_sum(const float* __restrict__ padded, const float* __restrict__ hmax,
                         float* __restrict__ hsum, int N, int H) {
    __shared__ float sd[256];
    int h = blockIdx.x;
    float m = hmax[h];
    float acc = 0.f;
    for (int n = threadIdx.x; n < N; n += blockDim.x) {
        float v = padded[n * H + h];
        acc += (v == -INFINITY) ? 0.f : expf(v - m);
    }
    sd[threadIdx.x] = acc; __syncthreads();
    for (int s = 128; s > 0; s >>= 1) {
        if (threadIdx.x < s) sd[threadIdx.x] += sd[threadIdx.x + s];
        __syncthreads();
    }
    if (threadIdx.x == 0) hsum[h] = sd[0];
}

// per-node softmax value (global softmax over node axis)
__global__ void sm_kernel(const float* __restrict__ padded, const float* __restrict__ hmax,
                          const float* __restrict__ hsum, float* __restrict__ sm,
                          int N, int H) {
    int i = blockIdx.x * blockDim.x + threadIdx.x;
    if (i >= N * H) return;
    int h = i % H;
    float v = padded[i];
    sm[i] = (v == -INFINITY) ? 0.f : expf(v - hmax[h]) / hsum[h];
}

// out[dst, h, f] += sm[dst, h] * Wh[src, h, f]   (vectorized over float4)
__global__ void scatter_kernel(const int* __restrict__ src, const int* __restrict__ dst,
                               const float* __restrict__ sm, const float* __restrict__ Wh,
                               float* __restrict__ out, int E, int H, int F) {
    unsigned int HF4 = (unsigned int)(H * F) / 4u;
    unsigned int i = blockIdx.x * blockDim.x + threadIdx.x;
    if (i >= (unsigned int)E * HF4) return;
    unsigned int e = i / HF4, j = i % HF4;
    int f4 = (int)j * 4;
    int h  = f4 / F;
    int s = src[e], d = dst[e];
    float w = sm[d * H + h];
    float4 x = *(const float4*)(Wh + (size_t)s * H * F + f4);
    float* o = out + (size_t)d * H * F + f4;
    atomicAdd(o + 0, w * x.x);
    atomicAdd(o + 1, w * x.y);
    atomicAdd(o + 2, w * x.z);
    atomicAdd(o + 3, w * x.w);
}

__global__ void elu_kernel(float* __restrict__ x, int n) {
    int i = blockIdx.x * blockDim.x + threadIdx.x;
    if (i < n) { float v = x[i]; x[i] = v > 0.f ? v : expm1f(v); }
}

// ---------------- host orchestration ----------------
static void gat_layer(const float* h_in, const unsigned short* Wt, const float* a,
                      int N, int E, const int* src, const int* dst,
                      int K, int H, int F,
                      unsigned short* hb, float* Wh, float* out,
                      float* as, float* ad, float* padded, float* smv,
                      float* hmax, float* hsum, hipStream_t stream) {
    const int Cols = H * F;
    const int nIn  = N * K;
    cvt_f32_bf16<<<(nIn + 255) / 256, 256, 0, stream>>>(h_in, hb, nIn);

    dim3 g((N + 127) / 128, Cols / 64);
    gemm_bf16_wmma<<<g, 256, 0, stream>>>((const bf16_t*)hb, (const bf16_t*)Wt, Wh, N, K, Cols);

    alpha_kernel<<<(N * H + 255) / 256, 256, 0, stream>>>(Wh, a, as, ad, N, H, F);
    fill_f32<<<(N * H + 255) / 256, 256, 0, stream>>>(padded, -INFINITY, N * H);
    score_kernel<<<(E * H + 255) / 256, 256, 0, stream>>>(src, dst, as, ad, padded, E, H);
    head_max<<<H, 256, 0, stream>>>(padded, hmax, N, H);
    head_sum<<<H, 256, 0, stream>>>(padded, hmax, hsum, N, H);
    sm_kernel<<<(N * H + 255) / 256, 256, 0, stream>>>(padded, hmax, hsum, smv, N, H);

    fill_f32<<<((N * Cols) + 255) / 256, 256, 0, stream>>>(out, 0.f, N * Cols);
    unsigned int total = (unsigned int)E * (unsigned int)(Cols / 4);
    scatter_kernel<<<(total + 255) / 256, 256, 0, stream>>>(src, dst, smv, Wh, out, E, H, F);
}

extern "C" void kernel_launch(void* const* d_in, const int* in_sizes, int n_in,
                              void* d_out, int out_size, void* d_ws, size_t ws_size,
                              hipStream_t stream) {
    const float* feat = (const float*)d_in[0];
    const int*   eidx = (const int*)d_in[1];
    const float* W1 = (const float*)d_in[2];
    const float* a1 = (const float*)d_in[3];
    const float* W2 = (const float*)d_in[4];
    const float* a2 = (const float*)d_in[5];
    const float* W3 = (const float*)d_in[6];
    const float* a3 = (const float*)d_in[7];

    const int N = in_sizes[0] / NODE_F;
    const int E = in_sizes[1] / 2;
    const int* src = eidx;
    const int* dst = eidx + E;
    const int HF = HEADS * HID;   // 512

    // workspace carve-up
    char* ws = (char*)d_ws;
    size_t off = 0;
    auto carve = [&](size_t bytes) -> void* {
        void* p = ws + off;
        off += (bytes + 255) & ~(size_t)255;
        return p;
    };
    float*          Wh    = (float*)carve((size_t)N * HF * sizeof(float));
    float*          hbuf  = (float*)carve((size_t)N * HF * sizeof(float));
    unsigned short* hb    = (unsigned short*)carve((size_t)N * HF * sizeof(unsigned short));
    unsigned short* W1t   = (unsigned short*)carve((size_t)NODE_F * HF * sizeof(unsigned short));
    unsigned short* W2t   = (unsigned short*)carve((size_t)HF * HF * sizeof(unsigned short));
    unsigned short* W3t   = (unsigned short*)carve((size_t)HF * OUTF * sizeof(unsigned short));
    float*          as    = (float*)carve((size_t)N * HEADS * sizeof(float));
    float*          ad    = (float*)carve((size_t)N * HEADS * sizeof(float));
    float*          padded= (float*)carve((size_t)N * HEADS * sizeof(float));
    float*          smv   = (float*)carve((size_t)N * HEADS * sizeof(float));
    float*          hmax  = (float*)carve(64 * sizeof(float));
    float*          hsum  = (float*)carve(64 * sizeof(float));
    (void)ws_size; (void)n_in;

    // pre-transpose + convert weights to bf16 (B^T layout for contiguous frags)
    {
        int n1 = NODE_F * HF, n2 = HF * HF, n3 = HF * OUTF;
        cvt_transpose_bf16<<<(n1 + 255) / 256, 256, 0, stream>>>(W1, W1t, NODE_F, HF);
        cvt_transpose_bf16<<<(n2 + 255) / 256, 256, 0, stream>>>(W2, W2t, HF, HF);
        cvt_transpose_bf16<<<(n3 + 255) / 256, 256, 0, stream>>>(W3, W3t, HF, OUTF);
    }

    // layer 1: feat [N,256] -> hbuf [N,512]
    gat_layer(feat, W1t, a1, N, E, src, dst, NODE_F, HEADS, HID,
              hb, Wh, hbuf, as, ad, padded, smv, hmax, hsum, stream);
    elu_kernel<<<((N * HF) + 255) / 256, 256, 0, stream>>>(hbuf, N * HF);

    // layer 2: hbuf [N,512] -> hbuf [N,512]  (in-place safe: hbuf consumed by cvt first)
    gat_layer(hbuf, W2t, a2, N, E, src, dst, HF, HEADS, HID,
              hb, Wh, hbuf, as, ad, padded, smv, hmax, hsum, stream);
    elu_kernel<<<((N * HF) + 255) / 256, 256, 0, stream>>>(hbuf, N * HF);

    // layer 3: hbuf [N,512] -> d_out [N,256], heads=1, no elu
    gat_layer(hbuf, W3t, a3, N, E, src, dst, HF, 1, OUTF,
              hb, Wh, (float*)d_out, as, ad, padded, smv, hmax, hsum, stream);
    (void)out_size;
}
